// ChainGNN_83391085019989
// MI455X (gfx1250) — compile-verified, matrix operand order checked by reading, not run on hardware
//
#include <hip/hip_runtime.h>

#define BSZ     64
#define ENCIN   64
#define ENCHID  256
#define NNODES  2048
#define GHD     128
#define NG      (NNODES * GHD)      // 262144
#define LN_EPS  1e-5f

typedef __attribute__((ext_vector_type(16))) __bf16 v16bf;
typedef __attribute__((ext_vector_type(8)))  float  v8f;

// CDNA5 16-bit A/B fragment K index for element j (half-word) in lane-half `khalf`
// (ISA 7.12.2: lanes 0-15: j0..7->K0..7, j8..15->K16..23; lanes 16-31: +8)
__device__ __forceinline__ int kmap(int j, int khalf) {
    return (j & 7) + ((j >> 3) << 4) + (khalf << 3);
}

// ---------------- encoder layer 1: H = relu(x@W1 + b1) -> bf16 ----------------
__global__ void k_enc1(const float* __restrict__ x, const float* __restrict__ W1,
                       const float* __restrict__ b1, __bf16* __restrict__ Hbf) {
    const int b = blockIdx.x;          // 0..63
    const int c = threadIdx.x;         // 0..255
    float acc = b1[c];
#pragma unroll 8
    for (int k = 0; k < ENCIN; ++k)
        acc = fmaf(x[b * ENCIN + k], W1[k * ENCHID + c], acc);
    Hbf[b * ENCHID + c] = (__bf16)fmaxf(acc, 0.f);
}

// ---------------- convert Wg1/Wg2 to bf16 ----------------
__global__ void k_cvt(const float* __restrict__ a, const float* __restrict__ b,
                      __bf16* __restrict__ oa, __bf16* __restrict__ ob, int n) {
    int i = blockIdx.x * blockDim.x + threadIdx.x;
    if (i < n) { oa[i] = (__bf16)a[i]; ob[i] = (__bf16)b[i]; }
}

// ---------------- encoder layer 2: out0 = H @ W2 + b2 (bf16 WMMA, f32 acc) ----------------
// One wave per 16-column tile; all 4 M-tiles share the streamed B fragment.
__global__ void __launch_bounds__(256) k_enc2(const __bf16* __restrict__ Hbf,
                                              const float* __restrict__ W2,
                                              const float* __restrict__ b2,
                                              float* __restrict__ out0) {
    const int lane    = threadIdx.x & 31;
    const int wave    = threadIdx.x >> 5;                 // 0..7
    const int colbase = (blockIdx.x * 8 + wave) * 16;     // column tile
    const int col     = colbase + (lane & 15);
    const int khalf   = lane >> 4;
    const int mrow    = lane & 15;

    v8f acc[4] = {};
    for (int kb = 0; kb < 8; ++kb) {                      // K = 256 = 8 * 32
        if (kb < 7)                                       // cover L2->WGP latency of the 268MB stream
            __builtin_prefetch(&W2[(size_t)((kb + 1) * 32 + khalf * 8) * NG + col], 0, 1);
        v16bf bfrag;
#pragma unroll
        for (int j = 0; j < 16; ++j) {
            int k = kb * 32 + kmap(j, khalf);
            bfrag[j] = (__bf16)W2[(size_t)k * NG + col];  // stream + native cvt
        }
#pragma unroll
        for (int mt = 0; mt < 4; ++mt) {
            v16bf afrag;
            int m = mt * 16 + mrow;
#pragma unroll
            for (int j = 0; j < 16; ++j) {
                int k = kb * 32 + kmap(j, khalf);
                afrag[j] = Hbf[m * ENCHID + k];
            }
            acc[mt] = __builtin_amdgcn_wmma_f32_16x16x32_bf16(
                false, afrag, false, bfrag, (short)0, acc[mt], false, false);
        }
    }
    const float bias = b2[col];
#pragma unroll
    for (int mt = 0; mt < 4; ++mt)
#pragma unroll
        for (int r = 0; r < 8; ++r) {
            int row = mt * 16 + r + khalf * 8;            // C/D layout: vgpr r, lane-half
            out0[(size_t)row * NG + col] = acc[mt][r] + bias;
        }
}

// ---------------- GCN half 1: Wx = LayerNorm(out0) @ Wg  (fused LN + WMMA) ----------------
__global__ void __launch_bounds__(256) k_lnmm(const float* __restrict__ out0,
                                              const float* __restrict__ g,
                                              const float* __restrict__ bb,
                                              const __bf16* __restrict__ Wgbf,
                                              float* __restrict__ Wx) {
    __shared__ __bf16 s_h[16][GHD + 8];                   // +8 pad: kills 16-way bank conflicts
    const int t  = threadIdx.x;
    const int r  = t >> 4;                                // row in tile 0..15
    const int c0 = t & 15;
    const size_t row = (size_t)blockIdx.x * 16 + r;

    float v[8], s = 0.f, s2 = 0.f;
#pragma unroll
    for (int j = 0; j < 8; ++j) {
        float f = out0[row * GHD + c0 + j * 16];
        v[j] = f; s += f; s2 += f * f;
    }
#pragma unroll
    for (int m = 8; m >= 1; m >>= 1) {                    // 16-lane tree reduce (wave32)
        s  += __shfl_xor(s,  m, 16);
        s2 += __shfl_xor(s2, m, 16);
    }
    const float mu   = s * (1.f / GHD);
    const float rstd = rsqrtf(s2 * (1.f / GHD) - mu * mu + LN_EPS);
#pragma unroll
    for (int j = 0; j < 8; ++j) {
        int c = c0 + j * 16;
        s_h[r][c] = (__bf16)((v[j] - mu) * rstd * g[c] + bb[c]);
    }
    __syncthreads();

    // GEMM: 8 waves, wave w -> columns [16w, 16w+16); K = 128 = 4 * 32
    const int lane  = t & 31;
    const int khalf = lane >> 4;
    const int m     = lane & 15;
    const int col   = (t >> 5) * 16 + (lane & 15);
    v8f acc = {};
#pragma unroll
    for (int kb = 0; kb < 4; ++kb) {
        v16bf afrag, bfrag;
#pragma unroll
        for (int j = 0; j < 16; ++j) {
            int k = kb * 32 + kmap(j, khalf);
            afrag[j] = s_h[m][k];
            bfrag[j] = Wgbf[k * GHD + col];
        }
        acc = __builtin_amdgcn_wmma_f32_16x16x32_bf16(
            false, afrag, false, bfrag, (short)0, acc, false, false);
    }
    const size_t rowbase = (size_t)blockIdx.x * 16;
#pragma unroll
    for (int rr = 0; rr < 8; ++rr)
        Wx[(rowbase + rr + khalf * 8) * GHD + col] = acc[rr];
}

// ---------------- GCN half 2: out0 += A_hat(tridiag) (x) Wx  (3-point stencil) ----------------
__global__ void k_stencil(float* __restrict__ out0, const float* __restrict__ Wx,
                          const float* __restrict__ A_hat) {
    const size_t idx = (size_t)blockIdx.x * blockDim.x + threadIdx.x;   // exact cover
    const int n = (int)((idx >> 7) & (NNODES - 1));
    float acc = out0[idx];
    if (n > 0)
        acc = fmaf(A_hat[(size_t)n * NNODES + (n - 1)], Wx[idx - GHD], acc);
    if (n < NNODES - 1)
        acc = fmaf(A_hat[(size_t)n * NNODES + (n + 1)], Wx[idx + GHD], acc);
    out0[idx] = acc;
}

// ---------------- head: y = out0 @ Wout + bout ----------------
__global__ void k_final(const float* __restrict__ out0, const float* __restrict__ Wout,
                        const float* __restrict__ bout, float* __restrict__ y) {
    const int t  = threadIdx.x;
    const int r  = t >> 4;
    const int c0 = t & 15;
    const size_t row = (size_t)blockIdx.x * 16 + r;
    float p = 0.f;
#pragma unroll
    for (int j = 0; j < 8; ++j) {
        int c = c0 + j * 16;
        p = fmaf(out0[row * GHD + c], Wout[c], p);
    }
#pragma unroll
    for (int m = 8; m >= 1; m >>= 1) p += __shfl_xor(p, m, 16);
    if (c0 == 0) y[row] = p + bout[0];
}

extern "C" void kernel_launch(void* const* d_in, const int* in_sizes, int n_in,
                              void* d_out, int out_size, void* d_ws, size_t ws_size,
                              hipStream_t stream) {
    (void)in_sizes; (void)n_in; (void)out_size; (void)ws_size;
    const float* x     = (const float*)d_in[0];
    const float* A_hat = (const float*)d_in[1];
    const float* W1    = (const float*)d_in[2];
    const float* b1    = (const float*)d_in[3];
    const float* W2    = (const float*)d_in[4];
    const float* b2    = (const float*)d_in[5];
    const float* Wg1   = (const float*)d_in[6];
    const float* Wg2   = (const float*)d_in[7];
    const float* g1    = (const float*)d_in[8];
    const float* lb1   = (const float*)d_in[9];
    const float* g2    = (const float*)d_in[10];
    const float* lb2   = (const float*)d_in[11];
    const float* Wout  = (const float*)d_in[12];
    const float* bout  = (const float*)d_in[13];

    char* ws = (char*)d_ws;
    float* out0 = (float*)ws;                                    // 64 MiB (B*N*GH f32)
    float* Wx   = (float*)(ws + ((size_t)64 << 20));             // 64 MiB
    __bf16* Hbf   = (__bf16*)(ws + ((size_t)128 << 20));         // 32 KB
    __bf16* Wg1bf = Hbf + BSZ * ENCHID;                          // 32 KB
    __bf16* Wg2bf = Wg1bf + GHD * GHD;                           // 32 KB

    k_enc1<<<BSZ, ENCHID, 0, stream>>>(x, W1, b1, Hbf);
    k_cvt<<<(GHD * GHD + 255) / 256, 256, 0, stream>>>(Wg1, Wg2, Wg1bf, Wg2bf, GHD * GHD);
    k_enc2<<<NG / 16 / 8, 256, 0, stream>>>(Hbf, W2, b2, out0);          // 2048 blocks

    k_lnmm<<<(BSZ * NNODES) / 16, 256, 0, stream>>>(out0, g1, lb1, Wg1bf, Wx);   // 8192
    k_stencil<<<(BSZ * NNODES * GHD) / 256, 256, 0, stream>>>(out0, Wx, A_hat);  // 65536

    k_lnmm<<<(BSZ * NNODES) / 16, 256, 0, stream>>>(out0, g2, lb2, Wg2bf, Wx);
    k_stencil<<<(BSZ * NNODES * GHD) / 256, 256, 0, stream>>>(out0, Wx, A_hat);

    k_final<<<(BSZ * NNODES) / 16, 256, 0, stream>>>(out0, Wout, bout, (float*)d_out);
}
